// TemporalFullAttention_44315472560742
// MI455X (gfx1250) — matrix operand order repaired and checked by reading, not compiled, not run on hardware
//
#include <hip/hip_runtime.h>

typedef __attribute__((ext_vector_type(16))) _Float16 v16h;
typedef __attribute__((ext_vector_type(8)))  _Float16 v8h;
typedef __attribute__((ext_vector_type(4)))  _Float16 v4h;
typedef __attribute__((ext_vector_type(2)))  _Float16 v2h;
typedef __attribute__((ext_vector_type(8)))  float    v8f;

#define B_  16
#define T_  1024
#define H_  8
#define DK_ 64

constexpr int KB  = 64;   // keys per iteration
constexpr int LKS = 80;   // ldsK row stride (halves): 160B, 16B-aligned rows
constexpr int LVS = 72;   // ldsVt row stride (halves): 144B, 16B-aligned rows
constexpr int LPS = 72;   // ldsP row stride (halves): 144B
constexpr int MW  = 2;    // query tiles (16 rows each) per wave
constexpr int NJ  = KB / 16;   // 4 score n-tiles per query tile
constexpr int NC  = KB / 32;   // 2 k-chunks for PV / P fragments

// one-instruction max (no NaN-canonicalize maxes)
__device__ __forceinline__ float vmax2(float a, float b) {
    float d;
    asm("v_max_num_f32_e32 %0, %1, %2" : "=v"(d) : "v"(a), "v"(b));
    return d;
}

// fused DPP reduction stage: d = max(lane_xor_N(x), x) in ONE v_max_num_f32_dpp
template<int XM> __device__ __forceinline__ float dpp_xmax(float x);
#define DPP_XMAX_DEF(N)                                                         \
template<> __device__ __forceinline__ float dpp_xmax<N>(float x) {              \
    float d;                                                                    \
    asm("v_max_num_f32_dpp %0, %1, %1 row_xmask:" #N                            \
        " row_mask:0xf bank_mask:0xf bound_ctrl:1"                              \
        : "=v"(d) : "v"(x));                                                    \
    return d; }
DPP_XMAX_DEF(1) DPP_XMAX_DEF(2) DPP_XMAX_DEF(4) DPP_XMAX_DEF(8)

__global__ __launch_bounds__(128) void fattn_fwd(const float* __restrict__ Q,
                                                 const float* __restrict__ K,
                                                 const float* __restrict__ V,
                                                 float* __restrict__ Out)
{
    __shared__ alignas(16) _Float16 ldsK [KB  * LKS];          // [64][80] keys row-major
    __shared__ alignas(16) _Float16 ldsVt[DK_ * LVS];          // [64][72] V transposed
    __shared__ alignas(16) _Float16 ldsP [4 * MW * 16 * LPS];  // per-wave P [32][72]

    const int tid  = threadIdx.x;
    const int wave = tid >> 5;
    const int lane = tid & 31;
    const int lm   = lane & 15;
    const int half = lane >> 4;

    const int qb = blockIdx.x;             // block of 128 queries
    const int h  = blockIdx.y;
    const int b  = blockIdx.z;
    const int q0 = qb * 128 + wave * 32;   // this wave's first query row (2 tiles)

    // ---- Q A-fragments, pre-scaled by (1/sqrt(64)) * log2(e): exp2 domain ----
    const float qscale = 0.125f * 1.4426950408889634f;
    v16h qa[MW][2];
    #pragma unroll
    for (int w2 = 0; w2 < MW; ++w2) {
        const float* qrow = Q + (((size_t)b * T_ + (q0 + w2 * 16 + lm)) * H_ + h) * DK_;
        #pragma unroll
        for (int c = 0; c < 2; ++c) {
            const int d0 = c * 32 + 8 * half;
            #pragma unroll
            for (int i = 0; i < 8; ++i) {
                qa[w2][c][i]     = (_Float16)(qrow[d0 + i]      * qscale);
                qa[w2][c][i + 8] = (_Float16)(qrow[d0 + 16 + i] * qscale);
            }
        }
    }

    v8f o_acc[MW][4], l_acc[MW], m_run[MW];
    #pragma unroll
    for (int w2 = 0; w2 < MW; ++w2) {
        l_acc[w2] = v8f{};
        #pragma unroll
        for (int t = 0; t < 4; ++t) o_acc[w2][t] = v8f{};
        #pragma unroll
        for (int v = 0; v < 8; ++v) m_run[w2][v] = -3.0e38f;
    }

    v16h ones;
    #pragma unroll
    for (int i = 0; i < 16; ++i) ones[i] = (_Float16)1.0f;

    _Float16* Pw = &ldsP[wave * (MW * 16) * LPS];

    for (int s0 = 0; s0 < T_; s0 += KB) {
        __syncthreads();   // previous iteration's readers done before overwrite

        // ---- cooperative K/V tile load: 64 keys x 64 dims, f32 -> f16 ----
        #pragma unroll
        for (int i = 0; i < 8; ++i) {
            const int e   = i * 128 + tid;   // float4 index (1024 total)
            const int row = e >> 4;
            const int c4  = (e & 15) * 4;
            const size_t goff = (((size_t)b * T_ + s0 + row) * H_ + h) * DK_ + c4;
            const float4 kk = *(const float4*)(K + goff);
            v4h kp; kp[0] = (_Float16)kk.x; kp[1] = (_Float16)kk.y;
                    kp[2] = (_Float16)kk.z; kp[3] = (_Float16)kk.w;
            *(v4h*)&ldsK[row * LKS + c4] = kp;   // one b64 store
            const float4 vv = *(const float4*)(V + goff);
            v2h va; va[0] = (_Float16)vv.x; va[1] = (_Float16)vv.y;  // cvt_pk
            v2h vb; vb[0] = (_Float16)vv.z; vb[1] = (_Float16)vv.w;  // cvt_pk
            ldsVt[(c4 + 0) * LVS + row] = va[0];
            ldsVt[(c4 + 1) * LVS + row] = va[1];
            ldsVt[(c4 + 2) * LVS + row] = vb[0];
            ldsVt[(c4 + 3) * LVS + row] = vb[1];
        }
        if (s0 + KB < T_) {   // uniform branch; emits global_prefetch_b8
            const int row = tid >> 1, c4 = (tid & 1) * 32;
            const size_t goff = (((size_t)b * T_ + s0 + KB + row) * H_ + h) * DK_ + c4;
            __builtin_prefetch(K + goff, 0, 1);
            __builtin_prefetch(V + goff, 0, 1);
        }
        __syncthreads();

        // ---- S = Q * K^T : 4 n-tiles; B-fragments shared across query tiles ----
        v8f s[MW][NJ];
        #pragma unroll
        for (int w2 = 0; w2 < MW; ++w2)
            #pragma unroll
            for (int j = 0; j < NJ; ++j) s[w2][j] = v8f{};
        #pragma unroll
        for (int j = 0; j < NJ; ++j) {
            #pragma unroll
            for (int c = 0; c < 2; ++c) {
                const _Float16* src = &ldsK[(j * 16 + lm) * LKS + c * 32 + 8 * half];
                const v8h lo = *(const v8h*)(src);
                const v8h hi = *(const v8h*)(src + 16);
                v16h bf;
                #pragma unroll
                for (int i = 0; i < 8; ++i) { bf[i] = lo[i]; bf[i + 8] = hi[i]; }
                #pragma unroll
                for (int w2 = 0; w2 < MW; ++w2)
                    s[w2][j] = __builtin_amdgcn_wmma_f32_16x16x32_f16(
                                   false, qa[w2][c], false, bf, (short)0, s[w2][j], false, false);
            }
        }

        // ---- online softmax in exp2 domain (one update per 64 keys) ----
        #pragma unroll
        for (int w2 = 0; w2 < MW; ++w2) {
            v8f mnv;
            #pragma unroll
            for (int v = 0; v < 8; ++v) {
                float mx = vmax2(vmax2(s[w2][0][v], s[w2][1][v]),
                                 vmax2(s[w2][2][v], s[w2][3][v]));
                mx = dpp_xmax<8>(dpp_xmax<4>(dpp_xmax<2>(dpp_xmax<1>(mx))));
                mnv[v] = vmax2(m_run[w2][v], mx);
            }
            const v8f dm = m_run[w2] - mnv;
            m_run[w2] = mnv;
            v8f av;
            #pragma unroll
            for (int v = 0; v < 8; ++v) av[v] = __builtin_amdgcn_exp2f(dm[v]);
            l_acc[w2] = l_acc[w2] * av;
            #pragma unroll
            for (int t = 0; t < 4; ++t) o_acc[w2][t] = o_acc[w2][t] * av;
            v8f d[NJ];
            #pragma unroll
            for (int j = 0; j < NJ; ++j) d[j] = s[w2][j] - mnv;
            #pragma unroll
            for (int v = 0; v < 8; ++v) {
                const int row = w2 * 16 + v + 8 * half;
                v2h hp01;                          // one v_cvt_pk_f16_f32 each
                hp01[0] = (_Float16)__builtin_amdgcn_exp2f(d[0][v]);
                hp01[1] = (_Float16)__builtin_amdgcn_exp2f(d[1][v]);
                v2h hp23;
                hp23[0] = (_Float16)__builtin_amdgcn_exp2f(d[2][v]);
                hp23[1] = (_Float16)__builtin_amdgcn_exp2f(d[3][v]);
                Pw[row * LPS + lm]      = hp01[0];
                Pw[row * LPS + 16 + lm] = hp01[1];   // d16_hi b16 store
                Pw[row * LPS + 32 + lm] = hp23[0];
                Pw[row * LPS + 48 + lm] = hp23[1];   // d16_hi b16 store
            }
        }
        asm volatile("s_wait_dscnt 0" ::: "memory");  // wave-private LDS RAW fence

        // ---- P A-fragments: [tile][k-chunk] ----
        v16h pf[MW][NC];
        #pragma unroll
        for (int w2 = 0; w2 < MW; ++w2) {
            const _Float16* pr = &Pw[(w2 * 16 + lm) * LPS];
            #pragma unroll
            for (int c = 0; c < NC; ++c) {
                const v8h lo = *(const v8h*)(pr + c * 32 + 8 * half);
                const v8h hi = *(const v8h*)(pr + c * 32 + 16 + 8 * half);
                #pragma unroll
                for (int i = 0; i < 8; ++i) { pf[w2][c][i] = lo[i]; pf[w2][c][i + 8] = hi[i]; }
            }
        }

        // ---- row sums: l += P x ones (WMMA instead of shuffle reduction) ----
        #pragma unroll
        for (int w2 = 0; w2 < MW; ++w2)
            #pragma unroll
            for (int c = 0; c < NC; ++c)
                l_acc[w2] = __builtin_amdgcn_wmma_f32_16x16x32_f16(
                                false, pf[w2][c], false, ones, (short)0, l_acc[w2], false, false);

        // ---- O += P * V : V fragments shared across both query tiles ----
        #pragma unroll
        for (int t = 0; t < 4; ++t) {
            #pragma unroll
            for (int c = 0; c < NC; ++c) {
                const _Float16* vr = &ldsVt[(t * 16 + lm) * LVS + c * 32 + 8 * half];
                const v8h lo = *(const v8h*)(vr);
                const v8h hi = *(const v8h*)(vr + 16);
                v16h vf;
                #pragma unroll
                for (int i = 0; i < 8; ++i) { vf[i] = lo[i]; vf[i + 8] = hi[i]; }
                #pragma unroll
                for (int w2 = 0; w2 < MW; ++w2)
                    o_acc[w2][t] = __builtin_amdgcn_wmma_f32_16x16x32_f16(
                                       false, pf[w2][c], false, vf, (short)0, o_acc[w2][t], false, false);
            }
        }
    }

    // ---- normalize and store [B, T, H, DK] fp32 ----
    #pragma unroll
    for (int w2 = 0; w2 < MW; ++w2) {
        #pragma unroll
        for (int v = 0; v < 8; ++v) {
            const float inv = 1.0f / l_acc[w2][v];
            const int q = q0 + w2 * 16 + v + 8 * half;
            float* orow = Out + (((size_t)b * T_ + q) * H_ + h) * DK_;
            #pragma unroll
            for (int t = 0; t < 4; ++t) orow[t * 16 + lm] = o_acc[w2][t][v] * inv;
        }
    }
}

extern "C" void kernel_launch(void* const* d_in, const int* in_sizes, int n_in,
                              void* d_out, int out_size, void* d_ws, size_t ws_size,
                              hipStream_t stream) {
    (void)in_sizes; (void)n_in; (void)out_size; (void)d_ws; (void)ws_size;
    const float* Q = (const float*)d_in[0];
    const float* K = (const float*)d_in[1];
    const float* V = (const float*)d_in[2];
    // d_in[3] = attention_mask (mask_flag=False -> unused)
    float* Out = (float*)d_out;

    dim3 grid(T_ / 128, H_, B_);   // 8 x 8 x 16 = 1024 blocks
    dim3 block(128);               // 4 waves; two 16-row query tiles per wave
    fattn_fwd<<<grid, block, 0, stream>>>(Q, K, V, Out);
}
